// SelfAttention_37469294690942
// MI455X (gfx1250) — matrix operand order repaired and checked by reading, not compiled, not run on hardware
//
#include <hip/hip_runtime.h>

// MI455X / gfx1250: wave32, WMMA 16x16x32 bf16 fp32-acc, async global->LDS
// double-buffered pipelines for both the projection GEMM and flash attention.

typedef __attribute__((ext_vector_type(16))) __bf16 v16bf;
typedef __attribute__((ext_vector_type(2)))  __bf16 v2bf;
typedef __attribute__((ext_vector_type(8)))  float  v8f;
typedef __attribute__((ext_vector_type(4)))  int    v4i;

#define B_  2
#define T_  2048
#define C_  1024
#define NH_ 16
#define HS_ 64

// ---------------- bf16 helpers ----------------
__device__ __forceinline__ unsigned short f2bf_bits(float f) {
  unsigned u = __builtin_bit_cast(unsigned, f);
  unsigned r = u + 0x7FFFu + ((u >> 16) & 1u);   // round-to-nearest-even
  return (unsigned short)(r >> 16);
}
__device__ __forceinline__ __bf16 bits2bf(unsigned short s) {
  return __builtin_bit_cast(__bf16, s);
}
__device__ __forceinline__ unsigned pack_bf2(float a, float b) {
#if __has_builtin(__builtin_amdgcn_cvt_pk_bf16_f32)
  v2bf r = __builtin_amdgcn_cvt_pk_bf16_f32(a, b);
  return __builtin_bit_cast(unsigned, r);
#else
  return (unsigned)f2bf_bits(a) | ((unsigned)f2bf_bits(b) << 16);
#endif
}
__device__ __forceinline__ v8f zero_v8f() {
  v8f z;
#pragma unroll
  for (int i = 0; i < 8; ++i) z[i] = 0.f;
  return z;
}

// ---------------- async global->LDS (present on this toolchain; sync fallback) ----
#if __has_builtin(__builtin_amdgcn_global_load_async_to_lds_b128)
#define HAVE_ASYNC 1
__device__ __forceinline__ void async_b128(const void* g, void* l) {
  __builtin_amdgcn_global_load_async_to_lds_b128(
      (__attribute__((address_space(1))) v4i*)(g),
      (__attribute__((address_space(3))) v4i*)(l),
      0, 0);
}
__device__ __forceinline__ void wait_async0() {
#if __has_builtin(__builtin_amdgcn_s_wait_asynccnt)
  __builtin_amdgcn_s_wait_asynccnt(0);
#else
  asm volatile("s_wait_asynccnt 0" ::: "memory");
#endif
}
#else
#define HAVE_ASYNC 0
#endif

// Tile loaders: ROWS x (CHUNKS*8) shorts, LDS row stride LST shorts (16B aligned).
#if HAVE_ASYNC
template <int ROWS, int CHUNKS, int NTH, int LST>
__device__ __forceinline__ void tile_load_async_t(const unsigned short* g, int gstride,
                                                  unsigned short (*lds)[LST], int tid) {
  constexpr int TOT = ROWS * CHUNKS;
#pragma unroll
  for (int e = 0; e < (TOT + NTH - 1) / NTH; ++e) {
    int idx = tid + e * NTH;
    if (TOT % NTH == 0 || idx < TOT) {
      int r = idx / CHUNKS, c = idx % CHUNKS;
      async_b128(g + (size_t)r * gstride + c * 8, &lds[r][c * 8]);
    }
  }
}
#endif
template <int ROWS, int DW, int NTH, int LST>   // DW = dwords per row
__device__ __forceinline__ void tile_load_sync_t(const unsigned short* g, int gstride,
                                                 unsigned short (*lds)[LST], int tid) {
  constexpr int TOT = ROWS * DW;
#pragma unroll
  for (int e = 0; e < (TOT + NTH - 1) / NTH; ++e) {
    int idx = tid + e * NTH;
    if (TOT % NTH == 0 || idx < TOT) {
      int r = idx / DW, c = idx % DW;
      *(unsigned*)&lds[r][c * 2] = *(const unsigned*)(g + (size_t)r * gstride + c * 2);
    }
  }
}

// ---------------------------------------------------------------------------
// Kernel A: pack x -> bf16 twice: xb row-major [B*T, C] (feeds projection GEMM)
// and qt transposed head-major [B*NH, HS, T] (feeds attention B-fragments).
// Tiled 64x64 transpose through LDS; grid (T/64, B*NH), 256 threads.
// ---------------------------------------------------------------------------
__global__ __launch_bounds__(256) void x_pack_kernel(
    const float* __restrict__ x, unsigned short* __restrict__ xb,
    unsigned short* __restrict__ qt) {
  __shared__ __align__(16) unsigned short sT[64][72];
  const int tid = threadIdx.x;
  const int bh = blockIdx.y, t0 = blockIdx.x * 64;
  const int b = bh >> 4, h = bh & 15;
#pragma unroll
  for (int e = 0; e < 8; ++e) {
    int idx = tid + e * 256;              // 2048 float-pairs
    int r = idx >> 5, c2 = idx & 31;      // r = t, c2 = hs pair
    size_t off = (size_t)(b * T_ + t0 + r) * C_ + h * HS_ + c2 * 2;
    unsigned v = pack_bf2(x[off], x[off + 1]);
    *(unsigned*)&sT[r][c2 * 2] = v;       // stage for transpose
    *(unsigned*)(xb + off) = v;           // row-major bf16 copy (coalesced)
  }
  __syncthreads();
#pragma unroll
  for (int e = 0; e < 8; ++e) {
    int idx = tid + e * 256;
    int r2 = idx >> 5, c2 = (idx & 31) * 2;   // r2 = hs, c2 = t pair
    unsigned v = (unsigned)sT[c2][r2] | ((unsigned)sT[c2 + 1][r2] << 16);
    *(unsigned*)(qt + ((size_t)bh * HS_ + r2) * T_ + t0 + c2) = v;
  }
}

// ---------------------------------------------------------------------------
// Kernel B: pack Wk/Wv -> bf16 (one-time, removes all conversion from GEMM loop).
// ---------------------------------------------------------------------------
__global__ __launch_bounds__(256) void w_pack_kernel(
    const float* __restrict__ Wk, const float* __restrict__ Wv,
    unsigned short* __restrict__ wkb, unsigned short* __restrict__ wvb) {
  const int n = C_ * C_ / 2;
  for (int i = blockIdx.x * blockDim.x + threadIdx.x; i < n;
       i += gridDim.x * blockDim.x) {
    size_t o = (size_t)i * 2;
    *(unsigned*)(wkb + o) = pack_bf2(Wk[o], Wk[o + 1]);
    *(unsigned*)(wvb + o) = pack_bf2(Wv[o], Wv[o + 1]);
  }
}

// ---------------------------------------------------------------------------
// Kernel C: fused K/V projection from bf16 inputs. 256 threads (8 waves),
// tile 128M x 32N, async double-buffered K-loop: the steady-state body is
// pure async_b128 issue -> ds_load_b128 fragments -> 4x WMMA.
// Output bf16 head-major [B*NH, T, HS], bias fused.
// ---------------------------------------------------------------------------
__global__ __launch_bounds__(256) void proj_kv_kernel(
    const unsigned short* __restrict__ xb, const unsigned short* __restrict__ wkb,
    const unsigned short* __restrict__ wvb,
    const float* __restrict__ bk, const float* __restrict__ bv,
    unsigned short* __restrict__ kp, unsigned short* __restrict__ vp) {
  __shared__ __align__(16) unsigned short sX[2][128][40];   // 128 x 32 (+pad)
  __shared__ __align__(16) unsigned short sWk[2][32][40];   // 32 x 32 (+pad)
  __shared__ __align__(16) unsigned short sWv[2][32][40];

  const int tid  = threadIdx.x;
  const int wave = tid >> 5;
  const int lane = tid & 31;
  const int lan  = lane & 15;
  const int kb   = (lane < 16) ? 0 : 8;
  const int m0   = blockIdx.y * 128;
  const int n0   = blockIdx.x * 32;

  const unsigned short* xrow = xb + (size_t)m0 * C_;

  v8f acc_k[2], acc_v[2];
#pragma unroll
  for (int nt = 0; nt < 2; ++nt) { acc_k[nt] = zero_v8f(); acc_v[nt] = zero_v8f(); }

#if HAVE_ASYNC
  tile_load_async_t<128, 4, 256, 40>(xrow, C_, sX[0], tid);
  tile_load_async_t<32, 4, 256, 40>(wkb + n0, C_, sWk[0], tid);
  tile_load_async_t<32, 4, 256, 40>(wvb + n0, C_, sWv[0], tid);
  wait_async0();
  __syncthreads();
#else
  tile_load_sync_t<128, 16, 256, 40>(xrow, C_, sX[0], tid);
  tile_load_sync_t<32, 16, 256, 40>(wkb + n0, C_, sWk[0], tid);
  tile_load_sync_t<32, 16, 256, 40>(wvb + n0, C_, sWv[0], tid);
  __syncthreads();
#endif

  int cur = 0;
  for (int kk = 0; kk < C_; kk += 32) {
    const int has_next = (kk + 32 < C_);
#if HAVE_ASYNC
    if (has_next) {
      tile_load_async_t<128, 4, 256, 40>(xrow + kk + 32, C_, sX[cur ^ 1], tid);
      tile_load_async_t<32, 4, 256, 40>(wkb + (size_t)(kk + 32) * C_ + n0, C_, sWk[cur ^ 1], tid);
      tile_load_async_t<32, 4, 256, 40>(wvb + (size_t)(kk + 32) * C_ + n0, C_, sWv[cur ^ 1], tid);
    }
#endif

    v16bf a;                              // A frag: row = wave*16+lan
    {
      const int row = wave * 16 + lan;
#pragma unroll
      for (int e = 0; e < 16; ++e) {
        int k = (e < 8) ? (kb + e) : (16 + kb + (e - 8));
        a[e] = bits2bf(sX[cur][row][k]);
      }
    }
#pragma unroll
    for (int nt = 0; nt < 2; ++nt) {
      v16bf bkf, bvf;                     // B frag: lane = N column, K in table order
#pragma unroll
      for (int e = 0; e < 16; ++e) {
        int k = (e < 8) ? (kb + e) : (16 + kb + (e - 8));
        bkf[e] = bits2bf(sWk[cur][k][nt * 16 + lan]);
        bvf[e] = bits2bf(sWv[cur][k][nt * 16 + lan]);
      }
      acc_k[nt] = __builtin_amdgcn_wmma_f32_16x16x32_bf16(false, a, false, bkf, (short)0, acc_k[nt], false, false);
      acc_v[nt] = __builtin_amdgcn_wmma_f32_16x16x32_bf16(false, a, false, bvf, (short)0, acc_v[nt], false, false);
    }

#if HAVE_ASYNC
    wait_async0();
    __syncthreads();
    cur ^= 1;
#else
    if (has_next) {
      __syncthreads();
      tile_load_sync_t<128, 16, 256, 40>(xrow + kk + 32, C_, sX[0], tid);
      tile_load_sync_t<32, 16, 256, 40>(wkb + (size_t)(kk + 32) * C_ + n0, C_, sWk[0], tid);
      tile_load_sync_t<32, 16, 256, 40>(wvb + (size_t)(kk + 32) * C_ + n0, C_, sWv[0], tid);
      __syncthreads();
    }
#endif
  }

  // C/D layout: VGPR r -> row (kb+r), lane -> column. Bias fused, head-major scatter.
#pragma unroll
  for (int nt = 0; nt < 2; ++nt) {
    const int gc = n0 + nt * 16 + lan;
    const int h  = gc >> 6, hs = gc & 63;
    const float bkv = bk[gc], bvv = bv[gc];
#pragma unroll
    for (int r = 0; r < 8; ++r) {
      int gm = m0 + wave * 16 + kb + r;
      int b  = gm >> 11, t = gm & 2047;
      size_t off = (((size_t)(b * NH_ + h) * T_ + t) * HS_ + hs);
      kp[off] = f2bf_bits(acc_k[nt][r] + bkv);
      vp[off] = f2bf_bits(acc_v[nt][r] + bvv);
    }
  }
}

// ---------------------------------------------------------------------------
// Kernel D: causal flash attention. Grid (T/64, B*NH), 128 threads (4 waves).
// Double-buffered Q/V tiles via async global->LDS (ASYNCcnt).
// scores[i][j] = k_i . q_j  (reference: "query" role is played by projected k).
// ---------------------------------------------------------------------------
__global__ __launch_bounds__(128) void flash_attn_kernel(
    const unsigned short* __restrict__ kp, const unsigned short* __restrict__ qt,
    const unsigned short* __restrict__ vp, float* __restrict__ out) {
  __shared__ __align__(16) unsigned short sK[64][72];
  __shared__ __align__(16) unsigned short sQ[2][64][72];   // [hs][j] (transposed)
  __shared__ __align__(16) unsigned short sV[2][64][72];   // [j][hs]
  __shared__ __align__(16) unsigned short sP[64][72];

  const int tid  = threadIdx.x;
  const int wave = tid >> 5;
  const int lane = tid & 31;
  const int lan  = lane & 15;
  const int kb   = (lane < 16) ? 0 : 8;
  const int i0   = blockIdx.x * 64;
  const int bh   = blockIdx.y;

  const unsigned short* Kb = kp + (size_t)bh * T_ * HS_;
  const unsigned short* Qb = qt + (size_t)bh * HS_ * T_;   // [HS][T]
  const unsigned short* Vb = vp + (size_t)bh * T_ * HS_;

#if HAVE_ASYNC
  tile_load_async_t<64, 8, 128, 72>(Kb + (size_t)i0 * HS_, HS_, sK, tid);
  tile_load_async_t<64, 8, 128, 72>(Qb, T_, sQ[0], tid);   // j0 = 0 columns
  tile_load_async_t<64, 8, 128, 72>(Vb, HS_, sV[0], tid);
  wait_async0();
  __syncthreads();
#else
  tile_load_sync_t<64, 32, 128, 72>(Kb + (size_t)i0 * HS_, HS_, sK, tid);
  tile_load_sync_t<64, 32, 128, 72>(Qb, T_, sQ[0], tid);
  tile_load_sync_t<64, 32, 128, 72>(Vb, HS_, sV[0], tid);
  __syncthreads();
#endif

  v8f accY[4];
#pragma unroll
  for (int f = 0; f < 4; ++f) accY[f] = zero_v8f();
  float mrow[8], lrow[8];
#pragma unroll
  for (int r = 0; r < 8; ++r) { mrow[r] = -1e30f; lrow[r] = 0.f; }

  const int arow = wave * 16 + lan;
  int cur = 0;

  for (int j0 = 0; j0 <= i0; j0 += 64) {
    const int has_next = (j0 + 64 <= i0);
#if HAVE_ASYNC
    if (has_next) {                        // prefetch next tile into other buffer
      tile_load_async_t<64, 8, 128, 72>(Qb + (j0 + 64), T_, sQ[cur ^ 1], tid);
      tile_load_async_t<64, 8, 128, 72>(Vb + (size_t)(j0 + 64) * HS_, HS_, sV[cur ^ 1], tid);
    }
#endif

    // ---- S = K_rows @ Q^T : WMMA-K = HS (2 steps of 32) ----
    v8f accS[4];
#pragma unroll
    for (int f = 0; f < 4; ++f) accS[f] = zero_v8f();
#pragma unroll
    for (int ks = 0; ks < 2; ++ks) {
      v16bf a;
#pragma unroll
      for (int e = 0; e < 16; ++e) {
        int k = ks * 32 + ((e < 8) ? (kb + e) : (16 + kb + (e - 8)));
        a[e] = bits2bf(sK[arow][k]);
      }
#pragma unroll
      for (int f = 0; f < 4; ++f) {
        v16bf bq;                          // B[k][n] = Qt[ks*32+k][j0+f*16+n] (k in table order)
#pragma unroll
        for (int e = 0; e < 16; ++e)
          bq[e] = bits2bf(sQ[cur][ks * 32 + ((e < 8) ? (kb + e) : (16 + kb + (e - 8)))][f * 16 + lan]);
        accS[f] = __builtin_amdgcn_wmma_f32_16x16x32_bf16(false, a, false, bq, (short)0, accS[f], false, false);
      }
    }

    // ---- causal mask (diagonal tile only) ----
    if (j0 == i0) {
#pragma unroll
      for (int f = 0; f < 4; ++f) {
        int jl = f * 16 + lan;
#pragma unroll
        for (int r = 0; r < 8; ++r) {
          int il = wave * 16 + kb + r;
          if (jl > il) accS[f][r] = -1e30f;
        }
      }
    }

    // ---- online softmax (C-frag: row = VGPR index, col = lane within half) ----
#pragma unroll
    for (int r = 0; r < 8; ++r) {
      float mx = accS[0][r];
#pragma unroll
      for (int f = 1; f < 4; ++f) mx = fmaxf(mx, accS[f][r]);
#pragma unroll
      for (int s = 1; s < 16; s <<= 1) mx = fmaxf(mx, __shfl_xor(mx, s, 32));
      float mnew  = fmaxf(mrow[r], mx);
      float alpha = __expf(mrow[r] - mnew);
      mrow[r] = mnew;
      float rs = 0.f;
#pragma unroll
      for (int f = 0; f < 4; ++f) {
        float p = __expf(accS[f][r] - mnew);
        accS[f][r] = p;
        rs += p;
      }
#pragma unroll
      for (int s = 1; s < 16; s <<= 1) rs += __shfl_xor(rs, s, 32);
      lrow[r] = lrow[r] * alpha + rs;
#pragma unroll
      for (int f = 0; f < 4; ++f) accY[f][r] *= alpha;
    }

    // ---- P -> LDS bf16 (C-frag -> A-frag relayout); wave touches only its rows ----
#pragma unroll
    for (int f = 0; f < 4; ++f) {
      int jl = f * 16 + lan;
#pragma unroll
      for (int r = 0; r < 8; ++r)
        sP[wave * 16 + kb + r][jl] = f2bf_bits(accS[f][r]);
    }

    // ---- Y += P @ V : WMMA-K = j (2 steps of 32) ----
#pragma unroll
    for (int ks = 0; ks < 2; ++ks) {
      v16bf a;
#pragma unroll
      for (int e = 0; e < 16; ++e) {
        int k = ks * 32 + ((e < 8) ? (kb + e) : (16 + kb + (e - 8)));
        a[e] = bits2bf(sP[arow][k]);
      }
#pragma unroll
      for (int f = 0; f < 4; ++f) {
        v16bf bvf;                         // B[k][n] = V[j0+ks*32+k][f*16+n] (k in table order)
#pragma unroll
        for (int e = 0; e < 16; ++e)
          bvf[e] = bits2bf(sV[cur][ks * 32 + ((e < 8) ? (kb + e) : (16 + kb + (e - 8)))][f * 16 + lan]);
        accY[f] = __builtin_amdgcn_wmma_f32_16x16x32_bf16(false, a, false, bvf, (short)0, accY[f], false, false);
      }
    }

#if HAVE_ASYNC
    wait_async0();                         // own wave's async loads done
    __syncthreads();                       // all waves' loads visible / reads done
    cur ^= 1;
#else
    if (has_next) {
      __syncthreads();
      tile_load_sync_t<64, 32, 128, 72>(Qb + (j0 + 64), T_, sQ[0], tid);
      tile_load_sync_t<64, 32, 128, 72>(Vb + (size_t)(j0 + 64) * HS_, HS_, sV[0], tid);
      __syncthreads();
    }
#endif
  }

  // ---- normalize, store fp32 [B, T, C] ----
  const int b = bh >> 4, h = bh & 15;
#pragma unroll
  for (int r = 0; r < 8; ++r) {
    int i = i0 + wave * 16 + kb + r;
    float inv = 1.0f / lrow[r];
#pragma unroll
    for (int f = 0; f < 4; ++f)
      out[(size_t)(b * T_ + i) * C_ + h * HS_ + f * 16 + lan] = accY[f][r] * inv;
  }
}

// ---------------------------------------------------------------------------
extern "C" void kernel_launch(void* const* d_in, const int* in_sizes, int n_in,
                              void* d_out, int out_size, void* d_ws, size_t ws_size,
                              hipStream_t stream) {
  const float* x  = (const float*)d_in[0];
  const float* Wk = (const float*)d_in[1];
  const float* bk = (const float*)d_in[2];
  const float* Wv = (const float*)d_in[3];
  const float* bv = (const float*)d_in[4];
  float* out = (float*)d_out;

  const size_t per = (size_t)B_ * NH_ * T_ * HS_;   // 4M shorts
  unsigned short* kp  = (unsigned short*)d_ws;
  unsigned short* vp  = kp + per;
  unsigned short* qt  = vp + per;
  unsigned short* xb  = qt + per;
  unsigned short* wkb = xb + per;
  unsigned short* wvb = wkb + (size_t)C_ * C_;

  dim3 gp(T_ / 64, B_ * NH_);
  x_pack_kernel<<<gp, 256, 0, stream>>>(x, xb, qt);

  w_pack_kernel<<<512, 256, 0, stream>>>(Wk, Wv, wkb, wvb);

  dim3 g1(C_ / 32, (B_ * T_) / 128);
  proj_kv_kernel<<<g1, 256, 0, stream>>>(xb, wkb, wvb, bk, bv, kp, vp);

  dim3 g3(T_ / 64, B_ * NH_);
  flash_attn_kernel<<<g3, 128, 0, stream>>>(kp, qt, vp, out);
}
